// GAT_68659347194336
// MI455X (gfx1250) — compile-verified, matrix operand order checked by reading, not run on hardware
//
#include <hip/hip_runtime.h>
#include <math.h>

// ---- problem constants (match reference) ----
#define NN     50000
#define EE     1600000
#define ETOT   (EE + NN)
#define NFEAT  512
#define H1     8
#define C1     16
#define F1     128   // H1*C1
#define C2     40
#define SLOPE  0.2f

typedef __attribute__((ext_vector_type(16))) _Float16 v16h;
typedef __attribute__((ext_vector_type(8)))  _Float16 v8h;
typedef __attribute__((ext_vector_type(8)))  float    v8f;

union FragAB { v16h v; v8h h[2]; };

__device__ __forceinline__ float lrelu(float x) { return x > 0.f ? x : SLOPE * x; }

// float atomic max via signed/unsigned integer ordering trick
__device__ __forceinline__ void atomicMaxF(float* addr, float v) {
  if (v >= 0.f) atomicMax((int*)addr, __float_as_int(v));
  else          atomicMin((unsigned int*)addr, __float_as_uint(v));
}

// edge e in [0, ETOT): first EE are real edges, rest are self loops
__device__ __forceinline__ void edge_sd(int e, const int* __restrict__ src,
                                        const int* __restrict__ dst, int& s, int& d) {
  if (e < EE) { s = src[e]; d = dst[e]; } else { s = e - EE; d = s; }
}

// ---------------- elementwise helpers ----------------
__global__ void cvt_f32_f16(const float* __restrict__ in, _Float16* __restrict__ out, int n) {
  for (int i = blockIdx.x * blockDim.x + threadIdx.x; i < n; i += gridDim.x * blockDim.x)
    out[i] = (_Float16)in[i];
}

__global__ void elu_to_f16(const float* __restrict__ in, _Float16* __restrict__ out, int n) {
  for (int i = blockIdx.x * blockDim.x + threadIdx.x; i < n; i += gridDim.x * blockDim.x) {
    float v = in[i];
    v = v > 0.f ? v : (__expf(v) - 1.f);
    out[i] = (_Float16)v;
  }
}

__global__ void fill_f32(float* __restrict__ p, float v, int n) {
  for (int i = blockIdx.x * blockDim.x + threadIdx.x; i < n; i += gridDim.x * blockDim.x)
    p[i] = v;
}

__global__ void fill_bias(float* __restrict__ p, const float* __restrict__ b, int C, int n) {
  for (int i = blockIdx.x * blockDim.x + threadIdx.x; i < n; i += gridDim.x * blockDim.x)
    p[i] = b[i % C];
}

// ---------------- WMMA GEMM: out[nrows x bcols] = A[nrows x K] * B[bcols x K]^T ----------------
// One wave computes a 16x16 tile via v_wmma_f32_16x16x32_f16.
// A fragment (16x32 f16): lanes 0-15 hold row M=lane, K=k0..k0+7 (v0..3) and k0+16..k0+23 (v4..7);
// lanes 16-31 hold K=k0+8.. and k0+24.. . B rows (output channels) use the same per-lane pattern.
template<int K, int CT, int RT>
__global__ __launch_bounds__(CT * RT * 32)
void gemm_wmma_f16(const _Float16* __restrict__ A, const _Float16* __restrict__ B,
                   float* __restrict__ out, int nrows, int bcols) {
  const int lane = threadIdx.x & 31;
  const int w    = threadIdx.x >> 5;
  const int rt   = w / CT, ct = w % CT;
  const int m0   = (blockIdx.x * RT + rt) * 16;
  if (m0 >= nrows) return;
  const int c0 = ct * 16;
  const int lr = lane & 15;
  const int hi = lane >> 4;

  const int arow = min(m0 + lr, nrows - 1);
  const int brow = min(c0 + lr, bcols - 1);
  const _Float16* pa = A + (size_t)arow * K + hi * 8;
  const _Float16* pb = B + (size_t)brow * K + hi * 8;

  v8f c = {};
#pragma unroll
  for (int k0 = 0; k0 < K; k0 += 32) {
    FragAB a, b;
    a.h[0] = *(const v8h*)(pa + k0);
    a.h[1] = *(const v8h*)(pa + k0 + 16);
    b.h[0] = *(const v8h*)(pb + k0);
    b.h[1] = *(const v8h*)(pb + k0 + 16);
    c = __builtin_amdgcn_wmma_f32_16x16x32_f16(false, a.v, false, b.v,
                                               (short)0, c, false, false);
  }

  const int n = c0 + lr;
  if (n < bcols) {
#pragma unroll
    for (int vi = 0; vi < 8; ++vi) {
      const int m = m0 + hi * 8 + vi;
      if (m < nrows) out[(size_t)m * bcols + n] = c[vi];
    }
  }
}

// ---------------- per-node attention halves: a = sum_c h[n,hd,c] * att[hd,c] ----------------
template<int H, int C, int HC>
__global__ void node_attn(const float* __restrict__ h, const float* __restrict__ attL,
                          const float* __restrict__ attR, float* __restrict__ aL,
                          float* __restrict__ aR) {
  int t = blockIdx.x * blockDim.x + threadIdx.x;
  if (t >= NN * H) return;
  int n = t / H, hd = t % H;
  const float* hp = h + (size_t)n * HC + hd * C;
  float sl = 0.f, sr = 0.f;
#pragma unroll
  for (int c = 0; c < C; ++c) {
    float v = hp[c];
    sl += v * attL[hd * C + c];
    sr += v * attR[hd * C + c];
  }
  aL[t] = sl; aR[t] = sr;
}

// ---------------- edge passes: segment max, segment denom ----------------
template<int H>
__global__ void edge_logit_max(const int* __restrict__ src, const int* __restrict__ dst,
                               const float* __restrict__ aL, const float* __restrict__ aR,
                               float* __restrict__ segmax) {
  long t = (long)blockIdx.x * blockDim.x + threadIdx.x;
  if (t >= (long)ETOT * H) return;
  int e = (int)(t / H), hd = (int)(t % H);
  int s, d; edge_sd(e, src, dst, s, d);
  float lg = lrelu(aL[s * H + hd] + aR[d * H + hd]);
  atomicMaxF(&segmax[d * H + hd], lg);
}

template<int H>
__global__ void edge_denom(const int* __restrict__ src, const int* __restrict__ dst,
                           const float* __restrict__ aL, const float* __restrict__ aR,
                           const float* __restrict__ segmax, float* __restrict__ denom) {
  long t = (long)blockIdx.x * blockDim.x + threadIdx.x;
  if (t >= (long)ETOT * H) return;
  int e = (int)(t / H), hd = (int)(t % H);
  int s, d; edge_sd(e, src, dst, s, d);
  float m = segmax[d * H + hd];
  if (m < -1e30f) m = 0.f;   // isfinite() guard from reference
  float lg = lrelu(aL[s * H + hd] + aR[d * H + hd]);
  atomicAdd(&denom[d * H + hd], __expf(lg - m));
}

// ---------------- weighted message scatter, layer 1: wave per edge, 4 channels/lane ----------------
__global__ __launch_bounds__(256)
void edge_scatter1(const int* __restrict__ src, const int* __restrict__ dst,
                   const float* __restrict__ ew, const float* __restrict__ aL,
                   const float* __restrict__ aR, const float* __restrict__ segmax,
                   const float* __restrict__ denom, const float* __restrict__ h,
                   float* __restrict__ acc) {
  int wid  = (blockIdx.x * blockDim.x + threadIdx.x) >> 5;
  int lane = threadIdx.x & 31;
  if (wid >= ETOT) return;
  int s, d; edge_sd(wid, src, dst, s, d);
  float w = ew[wid];
  int c  = lane * 4;        // 128 channels over 32 lanes
  int hd = c >> 4;          // head = channel / 16
  float m = segmax[d * H1 + hd];
  if (m < -1e30f) m = 0.f;
  float alpha = __expf(lrelu(aL[s * H1 + hd] + aR[d * H1 + hd]) - m) / denom[d * H1 + hd];
  float wa = w * alpha;
  const float4 hv = *(const float4*)(h + (size_t)s * F1 + c);
  float* ap = acc + (size_t)d * F1 + c;
  atomicAdd(ap + 0, wa * hv.x);
  atomicAdd(ap + 1, wa * hv.y);
  atomicAdd(ap + 2, wa * hv.z);
  atomicAdd(ap + 3, wa * hv.w);
}

// ---------------- weighted message scatter, layer 2: wave per edge, 40 channels ----------------
__global__ __launch_bounds__(256)
void edge_scatter2(const int* __restrict__ src, const int* __restrict__ dst,
                   const float* __restrict__ ew, const float* __restrict__ aL,
                   const float* __restrict__ aR, const float* __restrict__ segmax,
                   const float* __restrict__ denom, const float* __restrict__ h,
                   float* __restrict__ acc) {
  int wid  = (blockIdx.x * blockDim.x + threadIdx.x) >> 5;
  int lane = threadIdx.x & 31;
  if (wid >= ETOT) return;
  int s, d; edge_sd(wid, src, dst, s, d);
  float w = ew[wid];
  float m = segmax[d];
  if (m < -1e30f) m = 0.f;
  float alpha = __expf(lrelu(aL[s] + aR[d]) - m) / denom[d];
  float wa = w * alpha;
  atomicAdd(&acc[(size_t)d * C2 + lane], wa * h[(size_t)s * C2 + lane]);
  int c2 = lane + 32;
  if (c2 < C2)
    atomicAdd(&acc[(size_t)d * C2 + c2], wa * h[(size_t)s * C2 + c2]);
}

// ---------------- in-place row log-softmax over 40 classes: wave per node ----------------
__global__ __launch_bounds__(256)
void log_softmax40(float* __restrict__ io) {
  int wid  = (blockIdx.x * blockDim.x + threadIdx.x) >> 5;
  int lane = threadIdx.x & 31;
  if (wid >= NN) return;
  float* p = io + (size_t)wid * C2;
  float v0 = p[lane];
  bool  hv1 = (lane + 32) < C2;
  float v1 = hv1 ? p[lane + 32] : -__builtin_inff();
  float mx = fmaxf(v0, v1);
#pragma unroll
  for (int off = 16; off; off >>= 1) mx = fmaxf(mx, __shfl_xor(mx, off));
  float sm = __expf(v0 - mx) + (hv1 ? __expf(v1 - mx) : 0.f);
#pragma unroll
  for (int off = 16; off; off >>= 1) sm += __shfl_xor(sm, off);
  float ls = __logf(sm) + mx;
  p[lane] = v0 - ls;
  if (hv1) p[lane + 32] = v1 - ls;
}

// ---------------- host-side orchestration ----------------
extern "C" void kernel_launch(void* const* d_in, const int* in_sizes, int n_in,
                              void* d_out, int out_size, void* d_ws, size_t ws_size,
                              hipStream_t stream) {
  (void)in_sizes; (void)n_in; (void)out_size; (void)ws_size;
  const float* x     = (const float*)d_in[0];
  const int*   esrc  = (const int*)d_in[1];
  const int*   edst  = (const int*)d_in[2];
  const float* ew    = (const float*)d_in[3];
  const float* W1    = (const float*)d_in[4];
  const float* attL1 = (const float*)d_in[5];
  const float* attR1 = (const float*)d_in[6];
  const float* b1    = (const float*)d_in[7];
  const float* W2    = (const float*)d_in[8];
  const float* attL2 = (const float*)d_in[9];
  const float* attR2 = (const float*)d_in[10];
  const float* b2    = (const float*)d_in[11];
  float* out = (float*)d_out;

  // workspace carve-up (256B aligned)
  char* base = (char*)d_ws; size_t off = 0;
  auto alloc = [&](size_t bytes) -> void* {
    void* p = base + off;
    off = (off + bytes + 255) & ~(size_t)255;
    return p;
  };
  _Float16* x16 = (_Float16*)alloc((size_t)NN * NFEAT * 2);
  _Float16* w1h = (_Float16*)alloc((size_t)F1 * NFEAT * 2);
  float*    h1  = (float*)   alloc((size_t)NN * F1 * 4);
  float*    aL1 = (float*)   alloc((size_t)NN * H1 * 4);
  float*    aR1 = (float*)   alloc((size_t)NN * H1 * 4);
  float*    sm1 = (float*)   alloc((size_t)NN * H1 * 4);
  float*    dn1 = (float*)   alloc((size_t)NN * H1 * 4);
  float*    ac1 = (float*)   alloc((size_t)NN * F1 * 4);
  _Float16* h1e = (_Float16*)alloc((size_t)NN * F1 * 2);
  _Float16* w2h = (_Float16*)alloc((size_t)C2 * F1 * 2);
  float*    h2  = (float*)   alloc((size_t)NN * C2 * 4);
  float*    aL2 = (float*)   alloc((size_t)NN * 4);
  float*    aR2 = (float*)   alloc((size_t)NN * 4);
  float*    sm2 = (float*)   alloc((size_t)NN * 4);
  float*    dn2 = (float*)   alloc((size_t)NN * 4);

  const int TB = 256;
  auto cdiv = [](long a, long b) { return (int)((a + b - 1) / b); };
  const float NEG_INF = -__builtin_huge_valf();

  // ---- layer 1 ----
  cvt_f32_f16<<<4096, TB, 0, stream>>>(x, x16, NN * NFEAT);
  cvt_f32_f16<<<cdiv((long)F1 * NFEAT, TB), TB, 0, stream>>>(W1, w1h, F1 * NFEAT);
  gemm_wmma_f16<NFEAT, 8, 1><<<cdiv(NN, 16), 256, 0, stream>>>(x16, w1h, h1, NN, F1);
  node_attn<H1, C1, F1><<<cdiv((long)NN * H1, TB), TB, 0, stream>>>(h1, attL1, attR1, aL1, aR1);
  fill_f32<<<cdiv((long)NN * H1, TB), TB, 0, stream>>>(sm1, NEG_INF, NN * H1);
  fill_f32<<<cdiv((long)NN * H1, TB), TB, 0, stream>>>(dn1, 0.f, NN * H1);
  fill_bias<<<cdiv((long)NN * F1, TB), TB, 0, stream>>>(ac1, b1, F1, NN * F1);
  edge_logit_max<H1><<<cdiv((long)ETOT * H1, TB), TB, 0, stream>>>(esrc, edst, aL1, aR1, sm1);
  edge_denom<H1><<<cdiv((long)ETOT * H1, TB), TB, 0, stream>>>(esrc, edst, aL1, aR1, sm1, dn1);
  edge_scatter1<<<cdiv((long)ETOT * 32, TB), TB, 0, stream>>>(esrc, edst, ew, aL1, aR1, sm1, dn1, h1, ac1);
  elu_to_f16<<<cdiv((long)NN * F1, TB), TB, 0, stream>>>(ac1, h1e, NN * F1);

  // ---- layer 2 ----
  cvt_f32_f16<<<cdiv((long)C2 * F1, TB), TB, 0, stream>>>(W2, w2h, C2 * F1);
  gemm_wmma_f16<F1, 3, 2><<<cdiv(NN, 32), 192, 0, stream>>>(h1e, w2h, h2, NN, C2);
  node_attn<1, C2, C2><<<cdiv(NN, TB), TB, 0, stream>>>(h2, attL2, attR2, aL2, aR2);
  fill_f32<<<cdiv(NN, TB), TB, 0, stream>>>(sm2, NEG_INF, NN);
  fill_f32<<<cdiv(NN, TB), TB, 0, stream>>>(dn2, 0.f, NN);
  fill_bias<<<cdiv((long)NN * C2, TB), TB, 0, stream>>>(out, b2, C2, NN * C2);
  edge_logit_max<1><<<cdiv((long)ETOT, TB), TB, 0, stream>>>(esrc, edst, aL2, aR2, sm2);
  edge_denom<1><<<cdiv((long)ETOT, TB), TB, 0, stream>>>(esrc, edst, aL2, aR2, sm2, dn2);
  edge_scatter2<<<cdiv((long)ETOT * 32, TB), TB, 0, stream>>>(esrc, edst, ew, aL2, aR2, sm2, dn2, h2, out);
  log_softmax40<<<cdiv((long)NN * 32, TB), TB, 0, stream>>>(out);
}
